// MambaDecoderLayer_51281909514441
// MI455X (gfx1250) — matrix operand order repaired and checked by reading, not compiled, not run on hardware
//
#include <hip/hip_runtime.h>

// ---------------------------------------------------------------------------
// MambaDecoderLayer for MI455X (gfx1250, wave32, WMMA).
// GEMM operands staged in bf16 memory (halves streamed bytes, no cvt in the
// WMMA hot loop); accumulation f32 via v_wmma_f32_16x16x32_bf16.
// Elementwise math (conv, silu, softplus, scan, layernorm) stays fp32.
// ---------------------------------------------------------------------------

typedef __attribute__((ext_vector_type(16))) __bf16 v16bf;
typedef __attribute__((ext_vector_type(8)))  __bf16 bf16x8;
typedef __attribute__((ext_vector_type(4)))  __bf16 bf16x4;
typedef __attribute__((ext_vector_type(8)))  float  v8f;

#define LTGT 512
#define LMEM 512
#define BATCH 8
#define DMODEL 256
#define DIN 512
#define DSTATE 16
#define DTRANK 16

// ---------------- fp32 -> bf16 bulk convert (n % 4 == 0) --------------------
__global__ void __launch_bounds__(256)
f32_to_bf16_kernel(const float* __restrict__ in, __bf16* __restrict__ out, int n) {
  const int i = (blockIdx.x * blockDim.x + threadIdx.x) * 4;
  if (i >= n) return;
  const float4 v = *(const float4*)(in + i);
  bf16x4 o;
  o[0] = (__bf16)v.x; o[1] = (__bf16)v.y; o[2] = (__bf16)v.z; o[3] = (__bf16)v.w;
  *(bf16x4*)(out + i) = o;
}

// ---------------- generic GEMM: C(M,N) = A(M,K) @ W(N,K)^T [+bias][+relu] ---
// A, W are bf16 row-major. One wave computes a 16 x (16*NT) strip of C:
// one A fragment feeds NT WMMAs per 32-wide k-step.
// Requires K % 32 == 0, N % (16*NT) == 0, lda/ldw multiples of 16.
//
// WMMA operand layout as loaded here (wave32, contiguous k for A and B):
//  A 16x32 bf16: lanes 0-15 -> M=lane, K k0..k0+15;
//                lanes 16-31 -> M=lane-16, K k0+16..k0+31.
//  B 32x16 bf16 (from W(N,K) row-major, computing A @ W^T):
//                lane holds N = lane&15; lanes 0-15: K k0..k0+15,
//                lanes 16-31: K k0+16..k0+31  -> one aligned 32B load.
//  C/D v8f: element i at (M = (lane&16? 8:0)+i, N = lane&15).
template <int NT, bool BF16OUT>
__global__ void __launch_bounds__(128)
gemm_bf16_wmma(const __bf16* __restrict__ A, int lda,
               const __bf16* __restrict__ W, int ldw,
               void* __restrict__ Cv, int ldc, int M, int N, int K,
               const float* __restrict__ bias, int act) {
  const int lane   = threadIdx.x & 31;
  const int wave   = blockIdx.x * (blockDim.x >> 5) + (threadIdx.x >> 5);
  const int tilesN = N / (16 * NT);
  const int tm     = wave / tilesN;
  const int tn     = wave - tm * tilesN;
  if (tm * 16 >= M) return;   // wave-uniform: EXEC stays all-ones for WMMA

  const int r  = lane & 15;
  const int hi = (lane & 16) >> 1;                  // 0 or 8 (C/D row offset)
  const __bf16* ap  = A + (size_t)(tm * 16 + r) * lda + (lane & 16);
  const __bf16* wp0 = W + (size_t)(tn * NT * 16 + r) * ldw + (lane & 16);

  v8f acc[NT];
#pragma unroll
  for (int j = 0; j < NT; ++j) acc[j] = v8f{0.f,0.f,0.f,0.f,0.f,0.f,0.f,0.f};

  for (int k0 = 0; k0 < K; k0 += 32) {
    __builtin_prefetch(ap + k0 + 32, 0, 0);         // global_prefetch_b8
    const bf16x8 al = *(const bf16x8*)(ap + k0);
    const bf16x8 ah = *(const bf16x8*)(ap + k0 + 8);
    v16bf a;
#pragma unroll
    for (int e = 0; e < 8; ++e) { a[e] = al[e]; a[e + 8] = ah[e]; }
#pragma unroll
    for (int j = 0; j < NT; ++j) {
      const v16bf b = *(const v16bf*)(wp0 + (size_t)j * 16 * ldw + k0);
      acc[j] = __builtin_amdgcn_wmma_f32_16x16x32_bf16(
          /*neg_a=*/false, a, /*neg_b=*/false, b,
          /*c_mod=*/(short)0, acc[j], /*reuse_a=*/false, /*reuse_b=*/false);
    }
  }

  const int mbase = tm * 16 + hi;
#pragma unroll
  for (int j = 0; j < NT; ++j) {
    const int n = tn * NT * 16 + j * 16 + r;
    const float bv = bias ? bias[n] : 0.0f;
#pragma unroll
    for (int i = 0; i < 8; ++i) {
      float v = acc[j][i] + bv;
      if (act == 1) v = fmaxf(v, 0.0f);
      if (BF16OUT) ((__bf16*)Cv)[(size_t)(mbase + i) * ldc + n] = (__bf16)v;
      else         ((float*)Cv)[(size_t)(mbase + i) * ldc + n] = v;
    }
  }
}

// ---------------- dt projection: K=16, fused bias + softplus ----------------
// dtb(Ntok,512) = softplus(xdbl[:, :16] @ dt_proj_w(512,16)^T + dt_proj_b)
__global__ void __launch_bounds__(256)
dt_proj_kernel(const float* __restrict__ xdbl, const float* __restrict__ W,
               const float* __restrict__ bias, float* __restrict__ dtb,
               int Ntok) {
  const int i = blockIdx.x * blockDim.x + threadIdx.x;
  if (i >= Ntok * DIN) return;
  const int d   = i & (DIN - 1);
  const int tok = i >> 9;
  const float4* xq = (const float4*)(xdbl + (size_t)tok * 48);
  const float4* wq = (const float4*)(W + (size_t)d * DTRANK);
  float acc = bias[d];
#pragma unroll
  for (int q = 0; q < 4; ++q) {
    const float4 x = xq[q], w = wq[q];
    acc += x.x * w.x + x.y * w.y + x.z * w.z + x.w * w.w;
  }
  dtb[i] = (acc > 20.0f) ? acc : log1pf(__expf(acc));   // softplus
}

// ---------------- depthwise causal conv (k=4) + SiLU ------------------------
// x lives in xz rows (stride ldxz, columns 0..511).
// Writes fp32 (for the scan) and bf16 (A operand of the x_proj GEMM).
__global__ void __launch_bounds__(256)
conv_silu_kernel(const float* __restrict__ xz, int ldxz,
                 const float* __restrict__ w, const float* __restrict__ cb,
                 float* __restrict__ out, __bf16* __restrict__ out_bf,
                 int B, int L) {
  const int i = blockIdx.x * blockDim.x + threadIdx.x;
  if (i >= B * L * DIN) return;
  const int d   = i & (DIN - 1);
  const int tok = i >> 9;
  const int t   = tok % L;
  float acc = cb[d];
#pragma unroll
  for (int k = 0; k < 4; ++k) {
    const int tt = t - 3 + k;
    if (tt >= 0) acc += w[d * 4 + k] * xz[(size_t)(tok - 3 + k) * ldxz + d];
  }
  const float s = 1.0f / (1.0f + __expf(-acc));
  const float v = acc * s;                       // silu
  out[(size_t)tok * DIN + d]    = v;
  out_bf[(size_t)tok * DIN + d] = (__bf16)v;
}

// ---------------- selective scan: one lane per (batch, channel) -------------
// h[16] state in VGPRs (fp32). Emits bf16 y (A operand of out_proj GEMM).
__global__ void __launch_bounds__(256)
mamba_scan_kernel(const float* __restrict__ conv, const float* __restrict__ dtb,
                  const float* __restrict__ xdbl,
                  const float* __restrict__ xz, int ldxz,
                  const float* __restrict__ A_log, const float* __restrict__ Dp,
                  __bf16* __restrict__ yact, int B, int L) {
  const int i = blockIdx.x * blockDim.x + threadIdx.x;
  if (i >= B * DIN) return;
  const int d = i & (DIN - 1);
  const int b = i >> 9;
  float a[DSTATE], h[DSTATE];
#pragma unroll
  for (int n = 0; n < DSTATE; ++n) {
    a[n] = -__expf(A_log[d * DSTATE + n]);
    h[n] = 0.0f;
  }
  const float Dv = Dp[d];
  for (int t = 0; t < L; ++t) {
    const size_t tok = (size_t)b * L + t;
    const float dtv = dtb[tok * DIN + d];        // already softplus'ed
    const float xv  = conv[tok * DIN + d];
    const float zv  = xz[tok * ldxz + DIN + d];
    // xdbl row = [dt(16) | B(16) | C(16)], 48 floats, 16B aligned
    const float4* bq = (const float4*)(xdbl + tok * 48 + 16);
    float Bm[DSTATE], Cm[DSTATE];
#pragma unroll
    for (int q = 0; q < 4; ++q) {
      const float4 bb = bq[q], cc = bq[q + 4];
      Bm[4*q+0]=bb.x; Bm[4*q+1]=bb.y; Bm[4*q+2]=bb.z; Bm[4*q+3]=bb.w;
      Cm[4*q+0]=cc.x; Cm[4*q+1]=cc.y; Cm[4*q+2]=cc.z; Cm[4*q+3]=cc.w;
    }
    float y = 0.0f;
#pragma unroll
    for (int n = 0; n < DSTATE; ++n) {
      const float dA = __expf(dtv * a[n]);
      h[n] = dA * h[n] + dtv * Bm[n] * xv;
      y += h[n] * Cm[n];
    }
    const float sz = zv / (1.0f + __expf(-zv));  // silu(z)
    yact[tok * DIN + d] = (__bf16)((y + xv * Dv) * sz);
  }
}

// ---------------- layernorm over 256 features (one block per token) ---------
// mode 0: in row=row,          out row=row
// mode 1: in row=b*1024+t,     out row=row        (b=row>>9, t=row&511)
// mode 2: in row=row,          out row=t*8+b      (final transpose to (L,B,D))
// Optional bf16 mirror of the output (GEMM A-operand staging).
__global__ void __launch_bounds__(256)
ln_kernel(const float* __restrict__ x1, const float* __restrict__ x2,
          const float* __restrict__ g, const float* __restrict__ bta,
          float* __restrict__ out, __bf16* __restrict__ out_bf, int mode) {
  __shared__ float red[256];
  const int row = blockIdx.x;
  const int d   = threadIdx.x;
  const int b = row >> 9, t = row & 511;
  const size_t in1    = (mode == 1) ? ((size_t)b * 1024 + t) * DMODEL
                                    : (size_t)row * DMODEL;
  const size_t outoff = (mode == 2) ? ((size_t)t * BATCH + b) * DMODEL
                                    : (size_t)row * DMODEL;
  float v = x1[in1 + d];
  if (x2) v += x2[(size_t)row * DMODEL + d];
  red[d] = v; __syncthreads();
  for (int s = 128; s > 0; s >>= 1) { if (d < s) red[d] += red[d + s]; __syncthreads(); }
  const float mean = red[0] * (1.0f / 256.0f);
  __syncthreads();
  const float c = v - mean;
  red[d] = c * c; __syncthreads();
  for (int s = 128; s > 0; s >>= 1) { if (d < s) red[d] += red[d + s]; __syncthreads(); }
  const float var = red[0] * (1.0f / 256.0f);
  const float o = c * rsqrtf(var + 1e-5f) * g[d] + bta[d];
  out[outoff + d] = o;
  if (out_bf) out_bf[outoff + d] = (__bf16)o;
}

// ---------------- comb (8, 1024, 256) = [t_swapped ; mem_swapped] -> bf16 ---
__global__ void __launch_bounds__(256)
build_comb_kernel(const float* __restrict__ ln1, const float* __restrict__ mem,
                  __bf16* __restrict__ comb) {
  const int i = blockIdx.x * blockDim.x + threadIdx.x;
  if (i >= BATCH * (LTGT + LMEM) * DMODEL) return;
  const int d = i & (DMODEL - 1);
  const int t = (i >> 8) & 1023;
  const int b = i >> 18;
  const float v = (t < LTGT)
      ? ln1[((size_t)t * BATCH + b) * DMODEL + d]
      : mem[((size_t)(t - LTGT) * BATCH + b) * DMODEL + d];
  comb[i] = (__bf16)v;
}

// --------------------------------------------------------------------------
// Host-side orchestration
// --------------------------------------------------------------------------
static inline void cvt_bf16(const float* in, __bf16* out, int n, hipStream_t s) {
  f32_to_bf16_kernel<<<(n / 4 + 255) / 256, 256, 0, s>>>(in, out, n);
}

static inline void launch_gemm(const __bf16* A, int lda, const __bf16* W, int ldw,
                               void* C, int ldc, int M, int N, int K,
                               const float* bias, int act, bool bf16out,
                               hipStream_t s) {
  if (N % 64 == 0) {
    const int blocks = ((M / 16) * (N / 64) + 3) / 4;
    if (bf16out)
      gemm_bf16_wmma<4, true ><<<blocks, 128, 0, s>>>(A, lda, W, ldw, C, ldc, M, N, K, bias, act);
    else
      gemm_bf16_wmma<4, false><<<blocks, 128, 0, s>>>(A, lda, W, ldw, C, ldc, M, N, K, bias, act);
  } else {                                          // N == 48
    const int blocks = ((M / 16) * (N / 48) + 3) / 4;
    gemm_bf16_wmma<3, false><<<blocks, 128, 0, s>>>(A, lda, W, ldw, C, ldc, M, N, K, bias, act);
  }
}

// One full mamba mixer. u_bf: (B*L, 256) bf16 tokens.
// wbf[0..2] = in_proj_w, x_proj_w, out_proj_w (bf16);
// p[0..8]   = fp32 params (conv_w, conv_b at 1,2; dt at 4,5; A_log 6; D 7).
static void run_mamba(const __bf16* u_bf, int B, int L,
                      const float* const* p, const __bf16* const* wbf,
                      float* xz, float* conv, __bf16* conv_bf, float* xdbl,
                      float* dtb, __bf16* yact_bf, float* proj, hipStream_t s) {
  const int Ntok = B * L;
  // xz = u @ in_proj_w^T                             (Ntok, 1024) fp32
  launch_gemm(u_bf, DMODEL, wbf[0], DMODEL, xz, 2 * DIN, Ntok, 2 * DIN, DMODEL,
              nullptr, 0, false, s);
  // depthwise conv + silu                            (Ntok, 512) fp32+bf16
  conv_silu_kernel<<<(Ntok * DIN + 255) / 256, 256, 0, s>>>(
      xz, 2 * DIN, p[1], p[2], conv, conv_bf, B, L);
  // x_dbl = x @ x_proj_w^T                           (Ntok, 48) fp32
  launch_gemm(conv_bf, DIN, wbf[1], DIN, xdbl, 48, Ntok, 48, DIN,
              nullptr, 0, false, s);
  // dt = softplus(x_dbl[:, :16] @ dt_proj_w^T + b)   (Ntok, 512), K=16 -> FMA
  dt_proj_kernel<<<(Ntok * DIN + 255) / 256, 256, 0, s>>>(
      xdbl, p[4], p[5], dtb, Ntok);
  // selective scan (+ D skip + silu(z) gate) -> bf16 y
  mamba_scan_kernel<<<(B * DIN + 255) / 256, 256, 0, s>>>(
      conv, dtb, xdbl, xz, 2 * DIN, p[6], p[7], yact_bf, B, L);
  // out = y @ out_proj_w^T                           (Ntok, 256) fp32
  launch_gemm(yact_bf, DIN, wbf[2], DIN, proj, DMODEL, Ntok, DMODEL, DIN,
              nullptr, 0, false, s);
}

extern "C" void kernel_launch(void* const* d_in, const int* in_sizes, int n_in,
                              void* d_out, int out_size, void* d_ws, size_t ws_size,
                              hipStream_t stream) {
  (void)in_sizes; (void)n_in; (void)out_size; (void)ws_size;

  const float* tgt = (const float*)d_in[0];
  const float* mem = (const float*)d_in[1];
  const float* sm[9]; for (int i = 0; i < 9; ++i) sm[i] = (const float*)d_in[2 + i];
  const float* cm[9]; for (int i = 0; i < 9; ++i) cm[i] = (const float*)d_in[11 + i];
  const float* ln1_w = (const float*)d_in[20]; const float* ln1_b = (const float*)d_in[21];
  const float* ln2_w = (const float*)d_in[22]; const float* ln2_b = (const float*)d_in[23];
  const float* ln3_w = (const float*)d_in[24]; const float* ln3_b = (const float*)d_in[25];
  const float* lin1_w = (const float*)d_in[26]; const float* lin1_b = (const float*)d_in[27];
  const float* lin2_w = (const float*)d_in[28]; const float* lin2_b = (const float*)d_in[29];
  float* out = (float*)d_out;

  const int NT1 = LTGT * BATCH;                  // 4096 tokens (sm stage)
  const int NT2 = BATCH * (LTGT + LMEM);         // 8192 tokens (cm stage)

  // -------- workspace carve: fp32 region, then bf16 region -----------------
  float* ws   = (float*)d_ws;
  float* xz   = ws;                              // 8192*1024
  float* conv = xz   + (size_t)NT2 * 2 * DIN;    // 8192*512
  float* xdbl = conv + (size_t)NT2 * DIN;        // 8192*48
  float* dtb  = xdbl + (size_t)NT2 * 48;         // 8192*512
  float* proj = dtb  + (size_t)NT2 * DIN;        // 8192*256
  float* lnA  = proj + (size_t)NT2 * DMODEL;     // 4096*256 (ln1 out)
  float* lnB  = lnA  + (size_t)NT1 * DMODEL;     // 4096*256 (ln2 out)
  float* ffn2 = proj;                            // reuse after ln2 consumed it

  __bf16* bws     = (__bf16*)(lnB + (size_t)NT1 * DMODEL);
  __bf16* tgt_bf  = bws;                                  // 4096*256
  __bf16* comb_bf = tgt_bf  + (size_t)NT1 * DMODEL;       // 8192*256
  __bf16* conv_bf = comb_bf + (size_t)NT2 * DMODEL;       // 8192*512
  __bf16* yact_bf = conv_bf + (size_t)NT2 * DIN;          // 8192*512
  __bf16* lnB_bf  = yact_bf + (size_t)NT2 * DIN;          // 4096*256
  __bf16* ffn1_bf = lnB_bf  + (size_t)NT1 * DMODEL;       // 4096*1024
  __bf16* w_sm_in = ffn1_bf + (size_t)NT1 * 4 * DMODEL;   // 1024*256
  __bf16* w_sm_xp = w_sm_in + 1024 * 256;                 // 48*512
  __bf16* w_sm_op = w_sm_xp + 48 * 512;                   // 256*512
  __bf16* w_cm_in = w_sm_op + 256 * 512;
  __bf16* w_cm_xp = w_cm_in + 1024 * 256;
  __bf16* w_cm_op = w_cm_xp + 48 * 512;
  __bf16* w_lin1  = w_cm_op + 256 * 512;                  // 1024*256
  __bf16* w_lin2  = w_lin1  + 1024 * 256;                 // 256*1024

  // -------- stage 0: stage weights + tgt as bf16 ---------------------------
  cvt_bf16(tgt,    tgt_bf,  NT1 * DMODEL, stream);
  cvt_bf16(sm[0],  w_sm_in, 1024 * 256,   stream);
  cvt_bf16(sm[3],  w_sm_xp, 48 * 512,     stream);
  cvt_bf16(sm[8],  w_sm_op, 256 * 512,    stream);
  cvt_bf16(cm[0],  w_cm_in, 1024 * 256,   stream);
  cvt_bf16(cm[3],  w_cm_xp, 48 * 512,     stream);
  cvt_bf16(cm[8],  w_cm_op, 256 * 512,    stream);
  cvt_bf16(lin1_w, w_lin1,  1024 * 256,   stream);
  cvt_bf16(lin2_w, w_lin2,  256 * 1024,   stream);

  // -------- stage A: self mamba on tgt (batch=512, seq=8) ------------------
  const __bf16* sm_w[3] = { w_sm_in, w_sm_xp, w_sm_op };
  run_mamba(tgt_bf, LTGT, BATCH, sm, sm_w,
            xz, conv, conv_bf, xdbl, dtb, yact_bf, proj, stream);
  // t = LN1(tgt + mamba_out), layout stays (512, 8, 256)
  ln_kernel<<<NT1, 256, 0, stream>>>(proj, tgt, ln1_w, ln1_b, lnA, nullptr, 0);
  // comb (8, 1024, 256) bf16 = concat(t.swap01, memory.swap01)
  build_comb_kernel<<<(NT2 * DMODEL + 255) / 256, 256, 0, stream>>>(lnA, mem, comb_bf);

  // -------- stage B: cross mamba on comb (batch=8, seq=1024) ---------------
  const __bf16* cm_w[3] = { w_cm_in, w_cm_xp, w_cm_op };
  run_mamba(comb_bf, BATCH, LTGT + LMEM, cm, cm_w,
            xz, conv, conv_bf, xdbl, dtb, yact_bf, proj, stream);
  // t = LN2(mamba_out[:, :512, :]) -> (8, 512, 256) flat in lnB (+bf16)
  ln_kernel<<<NT1, 256, 0, stream>>>(proj, nullptr, ln2_w, ln2_b, lnB, lnB_bf, 1);

  // -------- FFN: relu(t @ lin1^T + b1) @ lin2^T + b2 -----------------------
  launch_gemm(lnB_bf, DMODEL, w_lin1, DMODEL, ffn1_bf, 4 * DMODEL,
              NT1, 4 * DMODEL, DMODEL, lin1_b, 1, /*bf16out=*/true, stream);
  launch_gemm(ffn1_bf, 4 * DMODEL, w_lin2, 4 * DMODEL, ffn2, DMODEL,
              NT1, DMODEL, 4 * DMODEL, lin2_b, 0, /*bf16out=*/false, stream);

  // -------- LN3(t + ff) with transpose back to (512, 8, 256) ---------------
  ln_kernel<<<NT1, 256, 0, stream>>>(lnB, ffn2, ln3_w, ln3_b, out, nullptr, 2);
}
